// SelfAttention_43069932044505
// MI455X (gfx1250) — compile-verified, matrix operand order checked by reading, not run on hardware
//
#include <hip/hip_runtime.h>
#include <hip/hip_bf16.h>
#include <stdint.h>

typedef __bf16 bf16;
typedef __attribute__((ext_vector_type(16))) __bf16 v16bf;
typedef __attribute__((ext_vector_type(8)))  float  v8f;
typedef __attribute__((ext_vector_type(4)))  int    v4i;

#if __has_builtin(__builtin_amdgcn_global_load_async_to_lds_b128) && \
    __has_builtin(__builtin_amdgcn_s_wait_asynccnt)
#define HAVE_ASYNC_LDS 1
#else
#define HAVE_ASYNC_LDS 0
#endif

union Frag {
  v16bf v;
  uint4 q[2];
  __bf16 e[16];
};

__device__ __forceinline__ v8f wmma_bf16(v16bf a, v16bf b, v8f c) {
  // D = A(16x32 bf16) x B(32x16 bf16) + C(16x16 f32)
  return __builtin_amdgcn_wmma_f32_16x16x32_bf16(false, a, false, b, (short)0, c, false, false);
}

#if HAVE_ASYNC_LDS
__device__ __forceinline__ void async_cp_b128(void* lds, const void* g) {
  __builtin_amdgcn_global_load_async_to_lds_b128(
      (__attribute__((address_space(1))) v4i*)(g),
      (__attribute__((address_space(3))) v4i*)(lds), 0, 0);
}
__device__ __forceinline__ void async_wait0() {
  __builtin_amdgcn_s_wait_asynccnt(0);
}
#endif

// ---------------------------------------------------------------- convert
__global__ void __launch_bounds__(256) cvt4_f32_bf16(const float4* __restrict__ in,
                                                     bf16* __restrict__ out, int n4) {
  int i = blockIdx.x * blockDim.x + threadIdx.x;
  if (i < n4) {
    float4 f = in[i];
    union { bf16 e[4]; uint2 u; } t;
    t.e[0] = (bf16)f.x; t.e[1] = (bf16)f.y; t.e[2] = (bf16)f.z; t.e[3] = (bf16)f.w;
    *(uint2*)(out + (size_t)i * 4) = t.u;
  }
}

// ---------------------------------------------------------------- GEMM out = X @ W^T
// X: MxK row-major bf16, W: NxK row-major bf16.
// MODE 0: bf16 out row-major (ldc = N)
// MODE 1: bf16 out scattered to (b,h,s,d) with S=2048,H=16,hd=64 (N must be 1024)
// MODE 2: f32 out row-major
#define GBM 128
#define GBK 32
#define XSTR 40     // bf16 elements per LDS row (80B, 16B-aligned, conflict-friendly)
#define GBUF 20480  // bytes per LDS buffer (Xs 10240 + Ws 10240)
#define GEMM_LDS (2 * GBUF)

template<int MODE>
__global__ void __launch_bounds__(256) gemm_xwt(const bf16* __restrict__ X,
                                                const bf16* __restrict__ W,
                                                void* __restrict__ Out,
                                                int M, int N, int K)
{
  extern __shared__ char smem[];
  // buffer bases computed from a runtime index (no constant pointer aggregates!)
  auto bufX = [&](int b) -> bf16* { return (bf16*)(smem + b * GBUF); };
  auto bufW = [&](int b) -> bf16* { return (bf16*)(smem + b * GBUF + 10240); };

  const int tid  = threadIdx.x;
  const int lane = tid & 31;
  const int w    = tid >> 5;
  const int ln   = lane & 15;
  const int hi   = lane >> 4;           // 0/1: K-group / row-half select
  const int wm   = (w & 3) * 32;        // wave row offset in 128-tile
  const int wn   = (w >> 2) * 64;       // wave col offset in 128-tile
  const int m0   = blockIdx.y * GBM;
  const int n0   = blockIdx.x * GBM;
  const int lrow = tid >> 1;
  const int lseg = tid & 1;

  v8f acc[2][4];
  const v8f vzero = {0.f,0.f,0.f,0.f,0.f,0.f,0.f,0.f};
  #pragma unroll
  for (int i = 0; i < 2; i++)
    #pragma unroll
    for (int j = 0; j < 4; j++) acc[i][j] = vzero;

  const int nk = K / GBK;

#if HAVE_ASYNC_LDS
  auto stage = [&](int kt, int b) {
    const bf16* sx = X + (size_t)(m0 + lrow) * K + kt * GBK + lseg * 16;
    const bf16* sw = W + (size_t)(n0 + lrow) * K + kt * GBK + lseg * 16;
    bf16* dx = bufX(b) + lrow * XSTR + lseg * 16;
    bf16* dw = bufW(b) + lrow * XSTR + lseg * 16;
    async_cp_b128(dx,     sx);
    async_cp_b128(dx + 8, sx + 8);
    async_cp_b128(dw,     sw);
    async_cp_b128(dw + 8, sw + 8);
  };
  stage(0, 0);
  async_wait0();
  __syncthreads();
#endif

  for (int kt = 0; kt < nk; kt++) {
#if HAVE_ASYNC_LDS
    const int cur = kt & 1;
    if (kt + 1 < nk) stage(kt + 1, (kt + 1) & 1);
#else
    const int cur = 0;
    __syncthreads();
    {
      const bf16* sx = X + (size_t)(m0 + lrow) * K + kt * GBK + lseg * 16;
      uint4* dx = (uint4*)(bufX(0) + lrow * XSTR + lseg * 16);
      dx[0] = ((const uint4*)sx)[0];
      dx[1] = ((const uint4*)sx)[1];
      const bf16* sw = W + (size_t)(n0 + lrow) * K + kt * GBK + lseg * 16;
      uint4* dw = (uint4*)(bufW(0) + lrow * XSTR + lseg * 16);
      dw[0] = ((const uint4*)sw)[0];
      dw[1] = ((const uint4*)sw)[1];
    }
    __syncthreads();
#endif

    const bf16* Xs = bufX(cur);
    const bf16* Ws = bufW(cur);
    Frag a[2], b[4];
    #pragma unroll
    for (int mi = 0; mi < 2; mi++) {
      const bf16* p = Xs + (wm + mi * 16 + ln) * XSTR + hi * 8;
      a[mi].q[0] = *(const uint4*)p;
      a[mi].q[1] = *(const uint4*)(p + 16);
    }
    #pragma unroll
    for (int ni = 0; ni < 4; ni++) {
      const bf16* p = Ws + (wn + ni * 16 + ln) * XSTR + hi * 8;
      b[ni].q[0] = *(const uint4*)p;
      b[ni].q[1] = *(const uint4*)(p + 16);
    }
    #pragma unroll
    for (int mi = 0; mi < 2; mi++)
      #pragma unroll
      for (int ni = 0; ni < 4; ni++)
        acc[mi][ni] = wmma_bf16(a[mi].v, b[ni].v, acc[mi][ni]);

#if HAVE_ASYNC_LDS
    async_wait0();
    __syncthreads();
#endif
  }

  #pragma unroll
  for (int mi = 0; mi < 2; mi++)
    #pragma unroll
    for (int ni = 0; ni < 4; ni++)
      #pragma unroll
      for (int i = 0; i < 8; i++) {
        int gr = m0 + wm + mi * 16 + hi * 8 + i;
        int gc = n0 + wn + ni * 16 + ln;
        float v = acc[mi][ni][i];
        if (MODE == 2) {
          ((float*)Out)[(size_t)gr * N + gc] = v;
        } else if (MODE == 1) {
          int bb = gr >> 11;   // S = 2048
          int ss = gr & 2047;
          int hh = gc >> 6;    // hd = 64
          int dd = gc & 63;
          ((bf16*)Out)[(((size_t)bb * 16 + hh) * 2048 + ss) * 64 + dd] = (bf16)v;
        } else {
          ((bf16*)Out)[(size_t)gr * N + gc] = (bf16)v;
        }
      }
}

// ---------------------------------------------------------------- flash attention
#define AS   2048
#define AHD  64
#define ABM  128
#define ABN  128
#define KSTR 72
#define VSTR 136
#define RSTR 264
#define SSTR 132
#define PSTR 136
#define OFF_VT  18432
#define OFF_RL  35840
#define OFF_SS  103424
#define OFF_PT  171008
#define OFF_MS  205824
#define OFF_LS  206336
#define OFF_FS  206848
#define ATT_LDS 207360

__global__ void __launch_bounds__(256) attn_kernel(const bf16* __restrict__ Q,
                                                   const bf16* __restrict__ Kb,
                                                   const bf16* __restrict__ Vb,
                                                   const bf16* __restrict__ Rel,
                                                   bf16* __restrict__ Y)
{
  extern __shared__ char smem[];
  bf16*  Ks  = (bf16*)(smem);              // [128][KSTR] key tile (row = key, col = d)
  bf16*  Vt  = (bf16*)(smem + OFF_VT);     // [64][VSTR]  value tile transposed (row = d)
  bf16*  Rl  = (bf16*)(smem + OFF_RL);     // [128][RSTR] rel-pos rows for this q block
  float* sS  = (float*)(smem + OFF_SS);    // [128][SSTR] logits scratch
  bf16*  Pt  = (bf16*)(smem + OFF_PT);     // [128][PSTR] softmax probabilities (bf16)
  float* mst = (float*)(smem + OFF_MS);    // running max per row
  float* lst = (float*)(smem + OFF_LS);    // running denom per row
  float* fst = (float*)(smem + OFF_FS);    // rescale factor per row

  const int tid  = threadIdx.x;
  const int lane = tid & 31;
  const int w    = tid >> 5;
  const int ln   = lane & 15;
  const int hi   = lane >> 4;
  const int bh   = blockIdx.y;             // b*16 + h
  const int i0   = blockIdx.x * ABM;
  const size_t baseq = (size_t)bh * AS;
  const int lrow = tid >> 1, lseg = tid & 1;

  // rel-pos tile: rows [i0, i0+128), all 256 relative positions
  {
    const bf16* src = Rel + ((size_t)bh * AS + i0 + lrow) * 256 + lseg * 128;
    bf16* dst = Rl + lrow * RSTR + lseg * 128;
    #pragma unroll
    for (int q = 0; q < 16; q++) {
#if HAVE_ASYNC_LDS
      async_cp_b128(dst + q * 8, src + q * 8);
#else
      ((uint4*)dst)[q] = ((const uint4*)src)[q];
#endif
    }
  }

  // q fragments in registers: wave owns query rows [i0 + w*16, +16), K = 64
  Frag qf[2];
  {
    const bf16* qp = Q + (baseq + i0 + w * 16 + ln) * AHD;
    #pragma unroll
    for (int f = 0; f < 2; f++) {
      qf[f].q[0] = *(const uint4*)(qp + f * 32 + hi * 8);
      qf[f].q[1] = *(const uint4*)(qp + f * 32 + 16 + hi * 8);
    }
  }

  if (tid < 128) { mst[tid] = -1e30f; lst[tid] = 0.f; fst[tid] = 1.f; }

  v8f ofr[4];
  const v8f vzero = {0.f,0.f,0.f,0.f,0.f,0.f,0.f,0.f};
  #pragma unroll
  for (int i = 0; i < 4; i++) ofr[i] = vzero;

  for (int kb = 0; kb < AS / ABN; kb++) {
    const int j0 = kb * ABN;
    __syncthreads();   // previous iteration's consumers of Ks/Vt/Pt are done
    { // K tile: 128 keys x 64 d
      const bf16* src = Kb + (baseq + j0 + lrow) * AHD + lseg * 32;
      bf16* dst = Ks + lrow * KSTR + lseg * 32;
      #pragma unroll
      for (int q = 0; q < 4; q++) {
#if HAVE_ASYNC_LDS
        async_cp_b128(dst + q * 8, src + q * 8);
#else
        ((uint4*)dst)[q] = ((const uint4*)src)[q];
#endif
      }
    }
    { // V tile transposed: Vt[d][key] (manual — needs element shuffle)
      const bf16* src = Vb + (baseq + j0 + lrow) * AHD + lseg * 32;
      #pragma unroll
      for (int q = 0; q < 4; q++) {
        union { uint4 u; bf16 e[8]; } t;
        t.u = ((const uint4*)src)[q];
        #pragma unroll
        for (int e = 0; e < 8; e++)
          Vt[(lseg * 32 + q * 8 + e) * VSTR + lrow] = t.e[e];
      }
    }
#if HAVE_ASYNC_LDS
    async_wait0();
#endif
    __syncthreads();

    // S = q @ k^T  (16 queries x 128 keys per wave), B-frags double-buffered
    {
      Frag bk[2][2];
      auto loadBK = [&](int nt, Frag* dst) {
        const bf16* p = Ks + (nt * 16 + ln) * KSTR + hi * 8;
        dst[0].q[0] = *(const uint4*)p;
        dst[0].q[1] = *(const uint4*)(p + 16);
        dst[1].q[0] = *(const uint4*)(p + 32);
        dst[1].q[1] = *(const uint4*)(p + 48);
      };
      loadBK(0, bk[0]);
      #pragma unroll
      for (int nt = 0; nt < 8; nt++) {
        if (nt < 7) loadBK(nt + 1, bk[(nt + 1) & 1]);
        v8f c = vzero;
        c = wmma_bf16(qf[0].v, bk[nt & 1][0].v, c);
        c = wmma_bf16(qf[1].v, bk[nt & 1][1].v, c);
        #pragma unroll
        for (int i = 0; i < 8; i++)
          sS[(w * 16 + hi * 8 + i) * SSTR + nt * 16 + ln] = c[i];
      }
    }
    __syncthreads();

    // online softmax: 2 lanes per query row, 64 keys each
    {
      const int r    = w * 16 + (lane >> 1);
      const int part = lane & 1;
      float* srow = sS + r * SSTR + part * 64;
      const bf16* rrow = Rl + r * RSTR;
      const int qglob = i0 + r;
      float bmax = -1e30f;
      for (int c = 0; c < 64; c++) {
        int j = j0 + part * 64 + c;
        int d = j - qglob;
        d = d < -128 ? -128 : (d > 127 ? 127 : d);
        float lg = (srow[c] + (float)rrow[d & 255]) * 0.125f;
        srow[c] = lg;
        bmax = fmaxf(bmax, lg);
      }
      bmax = fmaxf(bmax, __shfl_xor(bmax, 1, 32));
      float mo = mst[r];
      float mn = fmaxf(mo, bmax);
      float fac = __expf(mo - mn);
      float lsum = 0.f;
      bf16* prow = Pt + r * PSTR + part * 64;
      for (int c = 0; c < 64; c++) {
        float p = __expf(srow[c] - mn);
        prow[c] = (bf16)p;
        lsum += p;
      }
      lsum += __shfl_xor(lsum, 1, 32);
      if (part == 0) {
        mst[r] = mn;
        lst[r] = lst[r] * fac + lsum;
        fst[r] = fac;
      }
    }
    __syncthreads();

    // rescale running O, then O += P @ V (batched fragment loads per kk)
    {
      float fv[8];
      #pragma unroll
      for (int i = 0; i < 8; i++) fv[i] = fst[w * 16 + hi * 8 + i];
      #pragma unroll
      for (int dt = 0; dt < 4; dt++)
        #pragma unroll
        for (int i = 0; i < 8; i++) ofr[dt][i] *= fv[i];
    }
    #pragma unroll
    for (int kk = 0; kk < 4; kk++) {
      Frag af;
      const bf16* pp = Pt + (w * 16 + ln) * PSTR + kk * 32 + hi * 8;
      af.q[0] = *(const uint4*)pp;
      af.q[1] = *(const uint4*)(pp + 16);
      Frag bv[4];
      #pragma unroll
      for (int dt = 0; dt < 4; dt++) {
        const bf16* vp = Vt + (dt * 16 + ln) * VSTR + kk * 32 + hi * 8;
        bv[dt].q[0] = *(const uint4*)vp;
        bv[dt].q[1] = *(const uint4*)(vp + 16);
      }
      #pragma unroll
      for (int dt = 0; dt < 4; dt++)
        ofr[dt] = wmma_bf16(af.v, bv[dt].v, ofr[dt]);
    }
  }

  __syncthreads();
  { // normalize and write y in (b, s, e) layout for the final projection
    const int b = bh >> 4, h = bh & 15;
    float inv[8];
    #pragma unroll
    for (int i = 0; i < 8; i++) inv[i] = 1.f / lst[w * 16 + hi * 8 + i];
    #pragma unroll
    for (int dt = 0; dt < 4; dt++)
      #pragma unroll
      for (int i = 0; i < 8; i++) {
        int s = i0 + w * 16 + hi * 8 + i;
        Y[((size_t)b * AS + s) * 1024 + h * 64 + dt * 16 + ln] =
            (bf16)(ofr[dt][i] * inv[i]);
      }
  }
}

// ---------------------------------------------------------------- launcher
extern "C" void kernel_launch(void* const* d_in, const int* in_sizes, int n_in,
                              void* d_out, int out_size, void* d_ws, size_t ws_size,
                              hipStream_t stream)
{
  (void)in_sizes; (void)n_in; (void)out_size; (void)ws_size;
  const float* x   = (const float*)d_in[0];
  const float* pos = (const float*)d_in[1];
  const float* Wq  = (const float*)d_in[2];
  const float* Wk  = (const float*)d_in[3];
  const float* Wv  = (const float*)d_in[4];
  const float* Wp  = (const float*)d_in[5];
  float* out = (float*)d_out;

  char* ws = (char*)d_ws;
  size_t off = 0;
  auto alloc = [&](size_t bytes) -> char* {
    char* p = ws + off;
    off += (bytes + 255) & ~(size_t)255;
    return p;
  };

  const size_t NX = 2u * 2048u * 1024u;   // 4,194,304
  const size_t NW = 1024u * 1024u;        // 1,048,576
  const size_t NP = 256u * 64u;           // 16,384
  const size_t NR = 65536u * 256u;        // 16,777,216

  bf16* xb   = (bf16*)alloc(NX * 2);
  bf16* wqb  = (bf16*)alloc(NW * 2);
  bf16* wkb  = (bf16*)alloc(NW * 2);
  bf16* wvb  = (bf16*)alloc(NW * 2);
  bf16* wpb  = (bf16*)alloc(NW * 2);
  bf16* posb = (bf16*)alloc(NP * 2);
  bf16* qb   = (bf16*)alloc(NX * 2);
  bf16* kbf  = (bf16*)alloc(NX * 2);
  bf16* vbf  = (bf16*)alloc(NX * 2);
  bf16* relb = (bf16*)alloc(NR * 2);
  bf16* yb   = (bf16*)alloc(NX * 2);

  // fp32 -> bf16 conversions
  cvt4_f32_bf16<<<(int)(NX / 4 / 256), 256, 0, stream>>>((const float4*)x, xb, (int)(NX / 4));
  cvt4_f32_bf16<<<(int)(NW / 4 / 256), 256, 0, stream>>>((const float4*)Wq, wqb, (int)(NW / 4));
  cvt4_f32_bf16<<<(int)(NW / 4 / 256), 256, 0, stream>>>((const float4*)Wk, wkb, (int)(NW / 4));
  cvt4_f32_bf16<<<(int)(NW / 4 / 256), 256, 0, stream>>>((const float4*)Wv, wvb, (int)(NW / 4));
  cvt4_f32_bf16<<<(int)(NW / 4 / 256), 256, 0, stream>>>((const float4*)Wp, wpb, (int)(NW / 4));
  cvt4_f32_bf16<<<(int)(NP / 4 / 256), 256, 0, stream>>>((const float4*)pos, posb, (int)(NP / 4));

  // q/k/v projections: (4096x1024) @ (1024x1024)^T, scattered to (b,h,s,d)
  gemm_xwt<1><<<dim3(8, 32), 256, GEMM_LDS, stream>>>(xb, wqb, qb, 4096, 1024, 1024);
  gemm_xwt<1><<<dim3(8, 32), 256, GEMM_LDS, stream>>>(xb, wkb, kbf, 4096, 1024, 1024);
  gemm_xwt<1><<<dim3(8, 32), 256, GEMM_LDS, stream>>>(xb, wvb, vbf, 4096, 1024, 1024);

  // rel-pos logits: (b*h*s, 64) @ (256, 64)^T -> (b*h*s, 256) bf16
  gemm_xwt<0><<<dim3(2, 512), 256, GEMM_LDS, stream>>>(qb, posb, relb, 65536, 256, 64);

  // flash attention with rel-pos bias
  attn_kernel<<<dim3(16, 32), 256, ATT_LDS, stream>>>(qb, kbf, vbf, relb, yb);

  // output projection: (4096x1024) @ (1024x1024)^T -> fp32 out
  gemm_xwt<2><<<dim3(8, 32), 256, GEMM_LDS, stream>>>(yb, wpb, out, 4096, 1024, 1024);
}
